// VQNet_47656957116733
// MI455X (gfx1250) — matrix operand order repaired and checked by reading, not compiled, not run on hardware
//
#include <hip/hip_runtime.h>

#define N_TOK 16384
#define DIM   512
#define KCB   8192

#define TOK_PER_BLK    128        // tokens per block (8 waves x 16 tokens)
#define CODES_PER_TILE 16
#define NTILES         (KCB / CODES_PER_TILE)

typedef __attribute__((ext_vector_type(16))) __bf16 v16bf;
typedef __attribute__((ext_vector_type(8)))  __bf16 v8bf;
typedef __attribute__((ext_vector_type(8)))  float  v8f;

union V16 { v16bf v; v8bf h[2]; };

__device__ __forceinline__ void split_bf16(float f, __bf16& hi, __bf16& lo) {
    hi = (__bf16)f;              // round-to-nearest-even narrowing
    float hf = (float)hi;        // exact widening
    lo = (__bf16)(f - hf);
}

// ---------------------------------------------------------------------------
// CDNA5 async-copy helpers (ASYNCcnt-tracked global->LDS DMA, gfx1250).
// VDST operand carries the per-lane LDS byte address; 'off' = no SADDR (GV).
// ---------------------------------------------------------------------------
__device__ __forceinline__ void async_cp_b128(unsigned lds_off, const void* gsrc) {
    unsigned long long ga = (unsigned long long)(uintptr_t)gsrc;
    asm volatile("global_load_async_to_lds_b128 %0, %1, off"
                 :: "v"(lds_off), "v"(ga) : "memory");
}
__device__ __forceinline__ void wait_async_le8() {
    asm volatile("s_wait_asynccnt 0x8" ::: "memory");
}

// ---------------------------------------------------------------------------
// Kernel 1: e2[k] = ||embed[k]||^2 (exact fp32); optional bf16 hi/lo pre-split
// of the embed table into workspace; zero the loss slot.
// ---------------------------------------------------------------------------
__global__ void vq_prep(const float* __restrict__ e, float* __restrict__ e2,
                        __bf16* __restrict__ ehi_g, __bf16* __restrict__ elo_g,
                        float* __restrict__ loss_slot, int do_split) {
    int row = blockIdx.x;
    const float* r = e + (size_t)row * DIM;
    float s = 0.f;
    for (int i = threadIdx.x; i < DIM; i += blockDim.x) {
        float v = r[i];
        s += v * v;
        if (do_split) {
            __bf16 h, l;
            split_bf16(v, h, l);
            ehi_g[(size_t)row * DIM + i] = h;
            elo_g[(size_t)row * DIM + i] = l;
        }
    }
    for (int m = 16; m >= 1; m >>= 1) s += __shfl_xor(s, m, 32);
    __shared__ float red[8];
    int lane = threadIdx.x & 31, wv = threadIdx.x >> 5;
    if (lane == 0) red[wv] = s;
    __syncthreads();
    if (threadIdx.x == 0) {
        float t = 0.f;
        for (int i = 0; i < 8; ++i) t += red[i];
        e2[row] = t;
        if (row == 0) *loss_slot = 0.f;
    }
}

// ===========================================================================
// Kernel 2a (primary): async double-buffered WMMA argmin.
// LDS map: rows of 2048 B = { hi 1024 | lo 1024 }, XOR-swizzled 16B chunks
// within each 1024B half (chunk ^= row&15):
//   [0,      256K)  x tile: 128 rows x 2048 B
//   [256K,   320K)  e tile double buffer: 2 x (16 rows x 2048 B)
// Total = 327,680 B = full 320 KB WGP LDS.
// All fragment addresses = (loop-invariant lane base) + (kq*256 / +1024
// immediates folded into ds_load offsets).
// ===========================================================================
#define E_OFF   (256 * 1024)
#define EBUF_B  (CODES_PER_TILE * 2048)   // 32 KB per buffer
#define LDS_ASYNC_BYTES (E_OFF + 2 * EBUF_B)

// Each wave issues 8 async b128 copies per 32 KB e-tile (8 waves x 8 x 512B).
__device__ __forceinline__ void issue_etile(int tile, int buf, int wv, int lane,
                                            const __bf16* __restrict__ ehi_g,
                                            const __bf16* __restrict__ elo_g) {
    const int cb = tile * CODES_PER_TILE;
#pragma unroll
    for (int j = 0; j < 8; ++j) {
        int t   = wv * 256 + j * 32 + lane;   // 0..2047 transfer id (16B each)
        int row = t >> 7;                     // code row in tile (0..15)
        int q   = t & 127;                    // chunk within 2048B row
        int s   = q >> 6;                     // 0 = hi half, 1 = lo half
        int seg = q & 63;                     // 16B chunk within 1024B half
        const __bf16* g = (s ? elo_g : ehi_g) + (size_t)(cb + row) * DIM + seg * 8;
        unsigned lo = (unsigned)(E_OFF + buf * EBUF_B + (row << 11) + (s << 10)
                                 + ((seg ^ (row & 15)) << 4));
        async_cp_b128(lo, g);
    }
}

__global__ __launch_bounds__(256, 1)
void vq_argmin_async(const float* __restrict__ xs,
                     const __bf16* __restrict__ ehi_g,
                     const __bf16* __restrict__ elo_g,
                     const float* __restrict__ e2g,
                     int* __restrict__ idx_out) {
    extern __shared__ char smem[];
    const int tid     = threadIdx.x;
    const int tokbase = blockIdx.x * TOK_PER_BLK;

    // ---- Phase 1: load + split the 128x512 x tile, swizzled store ----
    for (int i = tid; i < TOK_PER_BLK * DIM; i += 256) {
        int r = i >> 9, c = i & (DIM - 1);
        float v = xs[(size_t)(tokbase + r) * DIM + c];
        __bf16 h, l;
        split_bf16(v, h, l);
        unsigned off = (unsigned)((r << 11) + (((c >> 3) ^ (r & 15)) << 4)
                                  + ((c & 7) << 1));
        *(__bf16*)(smem + off)        = h;   // hi half of row
        *(__bf16*)(smem + off + 1024) = l;   // lo half of row
    }

    const int lane = tid & 31;
    const int wv   = tid >> 5;
    const int m16  = lane & 15;
    const int hi16 = lane >> 4;
    const int arow = wv * 16 + m16;              // A row (token) for this lane
    const unsigned asw = (unsigned)(arow & 15);
    const unsigned bsw = (unsigned)m16;

    // Loop-invariant swizzled lane bases.
    // A chunk index: c = 16*kq + (4*km + hi16 + 2*w); phys = c ^ asw.
    // B chunk index: c = 16*kq + (4*km + hi16 + 2*w); phys = c ^ bsw.
    unsigned Abase[4][2], Binv[4][2];
#pragma unroll
    for (int km = 0; km < 4; ++km)
#pragma unroll
        for (int w = 0; w < 2; ++w) {
            Abase[km][w] = (unsigned)(arow << 11)
                         + ((((unsigned)(4 * km + hi16 + 2 * w) ^ asw) << 4));
            Binv[km][w]  = (unsigned)(m16 << 11)
                         + ((((unsigned)(4 * km + hi16 + 2 * w) ^ bsw) << 4));
        }

    float bestv[8]; int besti[8];
#pragma unroll
    for (int r = 0; r < 8; ++r) { bestv[r] = 3.4e38f; besti[r] = 0; }

    issue_etile(0, 0, wv, lane, ehi_g, elo_g);   // prologue: tile 0 -> buf 0

    for (int ct = 0; ct < NTILES; ++ct) {
        const int buf = ct & 1;
        // Prefetch next tile into the other buffer (wraps on last iter;
        // the extra copy is drained by S_ENDPGM's implicit wait-idle).
        issue_etile((ct + 1) & (NTILES - 1), buf ^ 1, wv, lane, ehi_g, elo_g);
        wait_async_le8();                        // tile ct's 8 copies done (in-order)
        __syncthreads();                         // all waves' copies visible

        const unsigned eb = (unsigned)(E_OFF + buf * EBUF_B);
        unsigned Bbase[4][2];
#pragma unroll
        for (int km = 0; km < 4; ++km) {
            Bbase[km][0] = eb + Binv[km][0];
            Bbase[km][1] = eb + Binv[km][1];
        }

        v8f acc0 = {0.f,0.f,0.f,0.f,0.f,0.f,0.f,0.f};
        v8f acc1 = acc0, acc2 = acc0;

#pragma unroll
        for (int kq = 0; kq < 4; ++kq) {
#pragma unroll
            for (int km = 0; km < 4; ++km) {
                const unsigned kqo = (unsigned)(kq << 8);   // folds to immediate
                V16 ah, al, bh, bl;
                ah.h[0] = *(const v8bf*)(smem + Abase[km][0] + kqo);
                ah.h[1] = *(const v8bf*)(smem + Abase[km][1] + kqo);
                al.h[0] = *(const v8bf*)(smem + Abase[km][0] + kqo + 1024);
                al.h[1] = *(const v8bf*)(smem + Abase[km][1] + kqo + 1024);
                bh.h[0] = *(const v8bf*)(smem + Bbase[km][0] + kqo);
                bh.h[1] = *(const v8bf*)(smem + Bbase[km][1] + kqo);
                bl.h[0] = *(const v8bf*)(smem + Bbase[km][0] + kqo + 1024);
                bl.h[1] = *(const v8bf*)(smem + Bbase[km][1] + kqo + 1024);

                acc0 = __builtin_amdgcn_wmma_f32_16x16x32_bf16(
                    false, ah.v, false, bh.v, (short)0, acc0, false, false);
                acc1 = __builtin_amdgcn_wmma_f32_16x16x32_bf16(
                    false, ah.v, false, bl.v, (short)0, acc1, false, false);
                acc2 = __builtin_amdgcn_wmma_f32_16x16x32_bf16(
                    false, al.v, false, bh.v, (short)0, acc2, false, false);
            }
        }

        const int   cb   = ct * CODES_PER_TILE;
        const float e2v  = e2g[cb + m16];        // 16 dwords, L2-resident
        const int   code = cb + m16;
#pragma unroll
        for (int r = 0; r < 8; ++r) {
            float dot  = acc0[r] + acc1[r] + acc2[r];
            float dist = e2v - 2.0f * dot;
            if (dist < bestv[r]) { bestv[r] = dist; besti[r] = code; }
        }
        __syncthreads();                         // readers done before overwrite
    }

    // ---- (min, argmin) across the 16 lanes sharing a token ----
#pragma unroll
    for (int r = 0; r < 8; ++r) {
        float v  = bestv[r];
        int   bi = besti[r];
        for (int m = 8; m >= 1; m >>= 1) {
            float ov = __shfl_xor(v, m, 32);
            int   oi = __shfl_xor(bi, m, 32);
            if (ov < v || (ov == v && oi < bi)) { v = ov; bi = oi; }
        }
        if (m16 == 0) idx_out[tokbase + wv * 16 + hi16 * 8 + r] = bi;
    }
}

// ===========================================================================
// Kernel 2b (fallback, small workspace): convert-in-kernel version.
// Padded rows (520 bf16), single-buffered e tile.
// ===========================================================================
#define ROW_BF 520
#define LDS_XTILE (TOK_PER_BLK * ROW_BF)
#define LDS_ETILE (CODES_PER_TILE * ROW_BF)
#define LDS_FB_BYTES (2*LDS_XTILE*2 + 2*LDS_ETILE*2 + CODES_PER_TILE*4)

__global__ __launch_bounds__(256, 1)
void vq_argmin(const float* __restrict__ xs, const float* __restrict__ e,
               const float* __restrict__ e2g, int* __restrict__ idx_out) {
    extern __shared__ char smem[];
    __bf16* xhi = (__bf16*)smem;
    __bf16* xlo = xhi + LDS_XTILE;
    __bf16* ehi = xlo + LDS_XTILE;
    __bf16* elo = ehi + LDS_ETILE;
    float*  e2t = (float*)(elo + LDS_ETILE);

    const int tid     = threadIdx.x;
    const int tokbase = blockIdx.x * TOK_PER_BLK;

    for (int i = tid; i < TOK_PER_BLK * DIM; i += 256) {
        int r = i >> 9, c = i & (DIM - 1);
        float v = xs[(size_t)(tokbase + r) * DIM + c];
        __bf16 h, l;
        split_bf16(v, h, l);
        xhi[r * ROW_BF + c] = h;
        xlo[r * ROW_BF + c] = l;
    }

    const int lane = tid & 31;
    const int wv   = tid >> 5;
    const int m16  = lane & 15;
    const int hi16 = lane >> 4;
    const int arow = wv * 16 + m16;

    float bestv[8]; int besti[8];
#pragma unroll
    for (int r = 0; r < 8; ++r) { bestv[r] = 3.4e38f; besti[r] = 0; }

    for (int ct = 0; ct < NTILES; ++ct) {
        const int cb = ct * CODES_PER_TILE;
        __syncthreads();
        for (int i = tid; i < CODES_PER_TILE * DIM; i += 256) {
            int r = i >> 9, c = i & (DIM - 1);
            float v = e[(size_t)(cb + r) * DIM + c];
            __bf16 h, l;
            split_bf16(v, h, l);
            ehi[r * ROW_BF + c] = h;
            elo[r * ROW_BF + c] = l;
        }
        if (tid < CODES_PER_TILE) e2t[tid] = e2g[cb + tid];
        __syncthreads();

        v8f acc0 = {0.f,0.f,0.f,0.f,0.f,0.f,0.f,0.f};
        v8f acc1 = acc0, acc2 = acc0;

#pragma unroll 4
        for (int kc = 0; kc < DIM / 32; ++kc) {
            const int d0 = kc * 32;
            const int ka = d0 + (hi16 << 3);
            V16 ah, al, bh, bl;
            const __bf16* ap  = &xhi[arow * ROW_BF + ka];
            ah.h[0] = *(const v8bf*)(ap);
            ah.h[1] = *(const v8bf*)(ap + 16);
            const __bf16* alp = &xlo[arow * ROW_BF + ka];
            al.h[0] = *(const v8bf*)(alp);
            al.h[1] = *(const v8bf*)(alp + 16);
            const int kb = d0 + (hi16 << 3);
            const __bf16* bp  = &ehi[m16 * ROW_BF + kb];
            bh.h[0] = *(const v8bf*)(bp);
            bh.h[1] = *(const v8bf*)(bp + 16);
            const __bf16* blp = &elo[m16 * ROW_BF + kb];
            bl.h[0] = *(const v8bf*)(blp);
            bl.h[1] = *(const v8bf*)(blp + 16);

            acc0 = __builtin_amdgcn_wmma_f32_16x16x32_bf16(
                false, ah.v, false, bh.v, (short)0, acc0, false, false);
            acc1 = __builtin_amdgcn_wmma_f32_16x16x32_bf16(
                false, ah.v, false, bl.v, (short)0, acc1, false, false);
            acc2 = __builtin_amdgcn_wmma_f32_16x16x32_bf16(
                false, al.v, false, bh.v, (short)0, acc2, false, false);
        }

        const float e2v  = e2t[m16];
        const int   code = cb + m16;
#pragma unroll
        for (int r = 0; r < 8; ++r) {
            float dot  = acc0[r] + acc1[r] + acc2[r];
            float dist = e2v - 2.0f * dot;
            if (dist < bestv[r]) { bestv[r] = dist; besti[r] = code; }
        }
    }

#pragma unroll
    for (int r = 0; r < 8; ++r) {
        float v  = bestv[r];
        int   bi = besti[r];
        for (int m = 8; m >= 1; m >>= 1) {
            float ov = __shfl_xor(v, m, 32);
            int   oi = __shfl_xor(bi, m, 32);
            if (ov < v || (ov == v && oi < bi)) { v = ov; bi = oi; }
        }
        if (m16 == 0) idx_out[tokbase + wv * 16 + hi16 * 8 + r] = bi;
    }
}

// ---------------------------------------------------------------------------
// Kernel 3: gather + straight-through output + loss.
// ---------------------------------------------------------------------------
__global__ void vq_gather_loss(const float* __restrict__ xs,
                               const float* __restrict__ e,
                               const int* __restrict__ idx,
                               float* __restrict__ out,
                               float* __restrict__ loss) {
    int token = blockIdx.x;
    int code  = idx[token];
    const float* xr   = xs  + (size_t)token * DIM;
    const float* er   = e   + (size_t)code  * DIM;
    float*       orow = out + (size_t)token * DIM;
    float s = 0.f;
    for (int i = threadIdx.x; i < DIM; i += blockDim.x) {
        float x = xr[i], ev = er[i];
        float d = ev - x;
        orow[i] = x + d;                 // matches reference xs + (out - xs)
        s += d * d;
    }
    for (int m = 16; m >= 1; m >>= 1) s += __shfl_xor(s, m, 32);
    __shared__ float red[8];
    int lane = threadIdx.x & 31, wv = threadIdx.x >> 5;
    if (lane == 0) red[wv] = s;
    __syncthreads();
    if (threadIdx.x == 0) {
        float t = 0.f;
        for (int i = 0; i < 8; ++i) t += red[i];
        atomicAdd(loss, 1.25f * t / (float)(N_TOK * DIM));
    }
}

// ---------------------------------------------------------------------------
extern "C" void kernel_launch(void* const* d_in, const int* in_sizes, int n_in,
                              void* d_out, int out_size, void* d_ws, size_t ws_size,
                              hipStream_t stream) {
    const float* xs = (const float*)d_in[0];   // [N_TOK, DIM]
    const float* ew = (const float*)d_in[1];   // [KCB, DIM]
    float* out  = (float*)d_out;               // [N_TOK*DIM] output_x, then [1] loss
    float* loss = out + (size_t)N_TOK * DIM;

    float*  e2    = (float*)d_ws;                              // 32 KB
    int*    idx   = (int*)(e2 + KCB);                          // 64 KB
    __bf16* ehi_g = (__bf16*)((char*)d_ws + 96 * 1024);        // 8 MB
    __bf16* elo_g = ehi_g + (size_t)KCB * DIM;                 // 8 MB

    const size_t need = 96 * 1024 + (size_t)2 * KCB * DIM * sizeof(__bf16);
    const int big = (ws_size >= need) ? 1 : 0;

    vq_prep<<<KCB, 256, 0, stream>>>(ew, e2, ehi_g, elo_g, loss, big);
    if (big) {
        vq_argmin_async<<<N_TOK / TOK_PER_BLK, 256, LDS_ASYNC_BYTES, stream>>>(
            xs, ehi_g, elo_g, e2, idx);
    } else {
        vq_argmin<<<N_TOK / TOK_PER_BLK, 256, LDS_FB_BYTES, stream>>>(
            xs, ew, e2, idx);
    }
    vq_gather_loss<<<N_TOK, 256, 0, stream>>>(xs, ew, idx, out, loss);
}
